// ImageClassifier_88244398064391
// MI455X (gfx1250) — matrix operand order repaired and checked by reading, not compiled, not run on hardware
//
#include <hip/hip_runtime.h>
#include <hip/hip_bf16.h>

// ---------------- problem constants ----------------
#define BATCH   64
#define FEAT    2048
#define NCH     112
#define TWO_E   1024
#define EHALF   512
#define NCLS    345
#define KHEAD   (NCH * EHALF)          // 57344
#define NTILE_H 22                     // ceil(345/16)
#define KSLICES 56                     // 57344 / 56 = 1024 K per slice
#define NPAD    352                    // 22*16

typedef __attribute__((ext_vector_type(16))) __bf16 v16bf;
typedef __attribute__((ext_vector_type(8)))  float  v8f;

struct U4    { unsigned x, y, z, w; };           // trivial 16B vector
struct B32x8 { unsigned u[8]; };                 // 8 dwords = one bf16 fragment
union  FragCvt { B32x8 b; v16bf v; };            // dword view <-> wmma operand view

__device__ __forceinline__ unsigned bf16_rne(float x) {
    unsigned u = __float_as_uint(x);
    return (u + 0x7FFFu + ((u >> 16) & 1u)) >> 16;   // round-to-nearest-even
}
__device__ __forceinline__ unsigned pack2(float lo, float hi) {
    return bf16_rne(lo) | (bf16_rne(hi) << 16);      // K even -> [15:0]
}

// ---------------- 1) x (f32) -> bf16 ----------------
__global__ void k_prep(const float* __restrict__ x, unsigned short* __restrict__ xb) {
    int i = blockIdx.x * blockDim.x + threadIdx.x;   // 64*2048 threads
    xb[i] = (unsigned short)bf16_rne(x[i]);
}

// ---------------- 2) embeddings GEMM (bf16 WMMA, W_emb read once) ----------------
__global__ void __launch_bounds__(256)
k_emb_gemm(const unsigned short* __restrict__ xb,
           const float* __restrict__ W_emb,
           const float* __restrict__ b_emb,
           const float* __restrict__ W_pint,
           float* __restrict__ emb,          // [b][c][e]  (64 x 112 x 1024)
           float* __restrict__ pint_part)    // [c][nt][b] (112 x 64 x 64)
{
    const int wave = (blockIdx.x << 3) + (threadIdx.x >> 5);   // 7168 waves
    const int c    = wave >> 6;            // 0..111
    const int nt   = wave & 63;            // 0..63  (16-wide N tile)
    const int n0   = nt << 4;
    const int lane = threadIdx.x & 31;
    const bool lo  = lane < 16;
    const int col  = lane & 15;
    const int n    = n0 + col;

    const U4* x4 = (const U4*)xb;

    v8f zero = {};
    v8f acc[4] = {zero, zero, zero, zero};   // M = 0..63 in 4 tiles of 16

    const float* wbase = W_emb + (size_t)c * FEAT * TWO_E + n;

    for (int k0 = 0; k0 < FEAT; k0 += 32) {
        // ---- B fragment: 32(K) x 16(N); lane = column n, lanes 16-31 hold K+16
        FragCvt bf;
        const float* wp = wbase + (size_t)(k0 + (lo ? 0 : 16)) * TWO_E;
        #pragma unroll
        for (int j = 0; j < 8; ++j)
            bf.b.u[j] = pack2(wp[(2 * j) * TWO_E], wp[(2 * j + 1) * TWO_E]);

        // ---- A fragments + WMMA
        #pragma unroll
        for (int t = 0; t < 4; ++t) {
            const int row = t * 16 + col;                       // M = row of x
            const U4* ap = x4 + ((row * FEAT + k0 + (lo ? 0 : 16)) >> 3);
            FragCvt af;
            U4 q0 = ap[0], q1 = ap[1];                          // K {0..7},{8..15} (lo) / +16 (hi)
            af.b.u[0] = q0.x; af.b.u[1] = q0.y; af.b.u[2] = q0.z; af.b.u[3] = q0.w;
            af.b.u[4] = q1.x; af.b.u[5] = q1.y; af.b.u[6] = q1.z; af.b.u[7] = q1.w;
            acc[t] = __builtin_amdgcn_wmma_f32_16x16x32_bf16(
                false, af.v, false, bf.v, (short)0, acc[t], false, false);
        }
    }

    // ---- epilogue: bias, store embeddings, partial W_pint dot (reduce over 16 lanes)
    const float bias = b_emb[c * TWO_E + n];
    const float wpv  = W_pint[c * TWO_E + n];
    #pragma unroll
    for (int t = 0; t < 4; ++t) {
        #pragma unroll
        for (int r = 0; r < 8; ++r) {
            const int b = t * 16 + r + (lo ? 0 : 8);            // D layout: VGPR r -> M=r / r+8
            float val = acc[t][r] + bias;
            emb[(size_t)b * (NCH * TWO_E) + c * TWO_E + n] = val;
            float s = val * wpv;
            s += __shfl_xor(s, 1);
            s += __shfl_xor(s, 2);
            s += __shfl_xor(s, 4);
            s += __shfl_xor(s, 8);                              // sum over the 16 columns
            if (col == 0)
                pint_part[((c << 6) + nt) * BATCH + b] = s;
        }
    }
}

// ---------------- 3) reduce pint partials + sigmoid ----------------
__global__ void k_pint(const float* __restrict__ pint_part,
                       const float* __restrict__ b_pint,
                       float* __restrict__ p_ws,
                       float* __restrict__ p_out)
{
    int i = blockIdx.x * blockDim.x + threadIdx.x;   // 64*112
    int b = i / NCH, c = i % NCH;
    float s = b_pint[c];
    const float* pp = pint_part + (size_t)(c << 6) * BATCH + b;
    #pragma unroll 8
    for (int nt = 0; nt < 64; ++nt) s += pp[nt * BATCH];
    float p = 1.0f / (1.0f + __expf(-s));
    p_ws[i]  = p;        // [b][c], for mixing
    p_out[i] = p;        // output #3
}

// ---------------- 4) mix -> flattened ----------------
__global__ void k_mix(const float* __restrict__ emb,
                      const float* __restrict__ p_ws,
                      float* __restrict__ flat)
{
    int i = blockIdx.x * blockDim.x + threadIdx.x;   // 64*112*512
    int e = i & 511;
    int c = (i >> 9) % NCH;
    int b = i / (NCH * EHALF);
    float p = p_ws[b * NCH + c];
    const float* eb = emb + (size_t)b * (NCH * TWO_E) + c * TWO_E;
    float pos = eb[e], neg = eb[EHALF + e];
    flat[(size_t)b * KHEAD + c * EHALF + e] = p * pos + (1.0f - p) * neg;
}

// ---------------- 5) head GEMM (bf16 WMMA, split-K, deterministic partials) ----------------
__global__ void __launch_bounds__(256)
k_head_gemm(const float* __restrict__ flat,
            const float* __restrict__ W_head,
            float* __restrict__ hpart)      // [ks][b][NPAD] (56 x 64 x 352)
{
    const int wave = (blockIdx.x << 3) + (threadIdx.x >> 5);   // 1232 waves
    const int nt   = wave % NTILE_H;        // 0..21
    const int ks   = wave / NTILE_H;        // 0..55
    const int lane = threadIdx.x & 31;
    const bool lo  = lane < 16;
    const int col  = lane & 15;
    const int n    = nt * 16 + col;
    const bool nv  = n < NCLS;

    v8f zero = {};
    v8f acc[4] = {zero, zero, zero, zero};

    const int kbeg = ks * (KHEAD / KSLICES);       // 1024 K per slice
    const int kend = kbeg + (KHEAD / KSLICES);

    for (int k0 = kbeg; k0 < kend; k0 += 32) {
        // ---- B fragment from W_head (row-major K x 345), zero-pad n >= 345
        FragCvt bf;
        {
            const int kb = k0 + (lo ? 0 : 16);
            const float* wp = W_head + (size_t)kb * NCLS + (nv ? n : 0);
            #pragma unroll
            for (int j = 0; j < 8; ++j) {
                float f0 = wp[(2 * j) * NCLS];
                float f1 = wp[(2 * j + 1) * NCLS];
                if (!nv) { f0 = 0.0f; f1 = 0.0f; }
                bf.b.u[j] = pack2(f0, f1);
            }
        }
        // ---- A fragments from flattened (f32 -> bf16 pairs)
        #pragma unroll
        for (int t = 0; t < 4; ++t) {
            const float2* ap =
                (const float2*)(flat + (size_t)(t * 16 + col) * KHEAD + k0 + (lo ? 0 : 16));
            FragCvt af;
            #pragma unroll
            for (int j = 0; j < 4; ++j) { float2 v = ap[j];     af.b.u[j]     = pack2(v.x, v.y); }
            #pragma unroll
            for (int j = 0; j < 4; ++j) { float2 v = ap[4 + j]; af.b.u[4 + j] = pack2(v.x, v.y); }
            acc[t] = __builtin_amdgcn_wmma_f32_16x16x32_bf16(
                false, af.v, false, bf.v, (short)0, acc[t], false, false);
        }
    }

    #pragma unroll
    for (int t = 0; t < 4; ++t) {
        #pragma unroll
        for (int r = 0; r < 8; ++r) {
            const int b = t * 16 + r + (lo ? 0 : 8);
            hpart[(size_t)(ks * BATCH + b) * NPAD + nt * 16 + col] = acc[t][r];
        }
    }
}

// ---------------- 6) reduce head partials + bias ----------------
__global__ void k_head_red(const float* __restrict__ hpart,
                           const float* __restrict__ b_head,
                           float* __restrict__ final_out)
{
    int i = blockIdx.x * blockDim.x + threadIdx.x;
    if (i >= BATCH * NCLS) return;
    int b = i / NCLS, n = i % NCLS;
    float s = b_head[n];
    const float* hp = hpart + (size_t)b * NPAD + n;
    #pragma unroll 8
    for (int ks = 0; ks < KSLICES; ++ks) s += hp[(size_t)ks * BATCH * NPAD];
    final_out[i] = s;
}

// ---------------- launcher ----------------
extern "C" void kernel_launch(void* const* d_in, const int* in_sizes, int n_in,
                              void* d_out, int out_size, void* d_ws, size_t ws_size,
                              hipStream_t stream) {
    (void)in_sizes; (void)n_in; (void)out_size; (void)ws_size;
    const float* x      = (const float*)d_in[0];
    const float* W_emb  = (const float*)d_in[1];
    const float* b_emb  = (const float*)d_in[2];
    const float* W_pint = (const float*)d_in[3];
    const float* b_pint = (const float*)d_in[4];
    const float* W_head = (const float*)d_in[5];
    const float* b_head = (const float*)d_in[6];

    float* out       = (float*)d_out;
    float* final_out = out;                                  // 64*345
    float* flat_out  = out + BATCH * NCLS;                   // 64*57344
    float* p_out     = out + BATCH * NCLS + BATCH * KHEAD;   // 64*112

    // workspace layout (bytes)
    char* ws = (char*)d_ws;
    unsigned short* xb   = (unsigned short*)(ws);                    // 262144 B
    float* p_ws          = (float*)(ws + 262144);                    // 28672 B
    float* emb           = (float*)(ws + 290816);                    // 29360128 B
    float* pint_part     = (float*)(ws + 29650944);                  // 1835008 B
    float* hpart         = (float*)(ws + 31485952);                  // 5046272 B

    k_prep     <<<512,   256, 0, stream>>>(x, xb);
    k_emb_gemm <<<896,   256, 0, stream>>>(xb, W_emb, b_emb, W_pint, emb, pint_part);
    k_pint     <<<28,    256, 0, stream>>>(pint_part, b_pint, p_ws, p_out);
    k_mix      <<<14336, 256, 0, stream>>>(emb, p_ws, flat_out);
    k_head_gemm<<<154,   256, 0, stream>>>(flat_out, W_head, hpart);
    k_head_red <<<87,    256, 0, stream>>>(hpart, b_head, final_out);
}
